// VQVAE_15238543966910
// MI455X (gfx1250) — compile-verified, matrix operand order checked by reading, not run on hardware
//
#include <hip/hip_runtime.h>

// ---------------------------------------------------------------------------
// VQ-VAE forward for MI455X (gfx1250, wave32, WMMA).
// Convs = implicit GEMM, v_wmma_f32_16x16x32_bf16, shape-specialized templates.
// 256-thread blocks (8 waves): A tile staged in LDS (bank-conflict-free,
// 80B row stride, ds_load_b128 fragments), B prepacked in WMMA fragment order
// (contiguous 32B per lane). Async global->LDS staging (ASYNCcnt) via the
// gfx1250 builtin (signature probed from compiler diagnostics).
// Activations NHWC bf16, f32 accumulate.
// ---------------------------------------------------------------------------

typedef __bf16 bf16;
typedef __attribute__((ext_vector_type(16))) __bf16 v16bf;
typedef __attribute__((ext_vector_type(8)))  __bf16 v8bf;
typedef __attribute__((ext_vector_type(8)))  float  v8f;
typedef int v4i_vs __attribute__((__vector_size__(16)));

#if __has_builtin(__builtin_amdgcn_global_load_async_to_lds_b128)
#define ATHENA_ASYNC_LDS 1
#else
#define ATHENA_ASYNC_LDS 0
#endif

__device__ __forceinline__ void wait_async0() {
#if __has_builtin(__builtin_amdgcn_s_wait_asynccnt)
  __builtin_amdgcn_s_wait_asynccnt(0);
#else
  asm volatile("s_wait_asynccnt 0" ::: "memory");
#endif
}

struct ConvP {
  const bf16*  in;      // NHWC bf16
  const bf16*  wpk;     // fragment-packed weights
  const float* bias;    // [Cout]
  bf16*        out_bf;  // NHWC bf16 (nullable)
  float*       out_f32; // nullable; out_mode 1=NHWC flat, 2=NCHW
  const bf16*  resid;   // NHWC bf16 (nullable), added before out_relu
  int Hin, Win, Hout, Wout, Cout;
  int wsh, hsh;         // log2(Wout), log2(Hout)
  int in_relu, out_relu, out_mode;
};

__device__ __forceinline__ unsigned relu2bf(unsigned x) {
  // packed 2x bf16 relu: zero each half whose sign bit is set
  unsigned lo = (x & 0x00008000u) ? 0u : (x & 0x0000ffffu);
  unsigned hi = (x & 0x80000000u) ? 0u : (x & 0xffff0000u);
  return lo | hi;
}

// One 256-thread block = 8 waves tiling (MW x NW) 16x16 WMMA tiles, MW = 8/NW.
template <int CIN, int KH, int KW, int STRIDE, int PAD, int TRN, int NW>
__global__ __launch_bounds__(256) void conv_wmma(ConvP p) {
  constexpr int MW    = 8 / NW;
  constexpr int MROWS = MW * 16;
  constexpr int K     = CIN * KH * KW;
  constexpr int NCH   = (K + 31) / 32;

  __shared__ bf16 lds_a[MROWS * 40];  // 80B row stride: conflict-free b128 reads

  const int tid  = threadIdx.x;
  const int lane = tid & 31;
  const int wv   = tid >> 5;
  const int wrow = wv / NW;
  const int wc   = wv % NW;
  const int grp  = lane >> 4;
  const int r16  = lane & 15;

  const int mbase  = blockIdx.x * MROWS;
  const int ntileG = blockIdx.y * NW + wc;
  const int ncol   = ntileG * 16 + r16;
  const int wmask  = p.Wout - 1;
  const int hmask  = p.Hout - 1;

  v8f acc = {};
  const v16bf* wf = (const v16bf*)p.wpk;
  const size_t wfbase = (size_t)ntileG * NCH;

  for (int c = 0; c < NCH; ++c) {
    const int k0 = c * 32;
    __syncthreads();

    if constexpr (CIN % 32 == 0) {
      // 32-wide K chunk = 32 consecutive ci at one (kh,kw) tap = 64B run
      const int cell = k0 / CIN;
      const int ci0  = k0 % CIN;
      const int kh   = cell / KW;
      const int kw   = cell % KW;
#if ATHENA_ASYNC_LDS
      if (!p.in_relu) {
        // async global->LDS (ASYNCcnt path); zero-fill holes via ds store
        for (int t = tid; t < MROWS * 4; t += 256) {
          const int row = t >> 2;
          const int q   = t & 3;
          const int m   = mbase + row;
          const int ox  = m & wmask;
          const int oy  = (m >> p.wsh) & hmask;
          const int b   = m >> (p.wsh + p.hsh);
          int iy, ix;
          bool ok;
          if constexpr (TRN) {
            const int dy = oy + kh - 2;
            const int dx = ox + kw - 2;
            ok = (dy >= 0) && (dx >= 0) && ((dy & 1) == 0) && ((dx & 1) == 0);
            iy = dy >> 1;
            ix = dx >> 1;
            ok = ok && (iy < p.Hin) && (ix < p.Win);
          } else {
            iy = oy * STRIDE - PAD + kh;
            ix = ox * STRIDE - PAD + kw;
            ok = (iy >= 0) && (iy < p.Hin) && (ix >= 0) && (ix < p.Win);
          }
          bf16* dst = &lds_a[row * 40 + q * 8];
          if (ok) {
            const bf16* src = p.in +
                ((size_t)(b * p.Hin + iy) * p.Win + ix) * CIN + ci0 + q * 8;
            __builtin_amdgcn_global_load_async_to_lds_b128(
                (__attribute__((address_space(1))) v4i_vs*)src,
                (__attribute__((address_space(3))) v4i_vs*)dst, 0, 0);
          } else {
            uint4 z = {0u, 0u, 0u, 0u};
            *(uint4*)dst = z;
          }
        }
        wait_async0();
      } else
#endif
      {
        for (int t = tid; t < MROWS * 4; t += 256) {
          const int row = t >> 2;
          const int q   = t & 3;
          const int m   = mbase + row;
          const int ox  = m & wmask;
          const int oy  = (m >> p.wsh) & hmask;
          const int b   = m >> (p.wsh + p.hsh);
          int iy, ix;
          bool ok;
          if constexpr (TRN) {  // lhs_dilation=2, pad=2 (convT k4 s2 p1)
            const int dy = oy + kh - 2;
            const int dx = ox + kw - 2;
            ok = (dy >= 0) && (dx >= 0) && ((dy & 1) == 0) && ((dx & 1) == 0);
            iy = dy >> 1;
            ix = dx >> 1;
            ok = ok && (iy < p.Hin) && (ix < p.Win);
          } else {
            iy = oy * STRIDE - PAD + kh;
            ix = ox * STRIDE - PAD + kw;
            ok = (iy >= 0) && (iy < p.Hin) && (ix >= 0) && (ix < p.Win);
          }
          uint4 val = {0u, 0u, 0u, 0u};
          if (ok) {
            const uint4* src = (const uint4*)(p.in +
                ((size_t)(b * p.Hin + iy) * p.Win + ix) * CIN + ci0 + q * 8);
            val = *src;
            if (p.in_relu) {
              val.x = relu2bf(val.x); val.y = relu2bf(val.y);
              val.z = relu2bf(val.z); val.w = relu2bf(val.w);
            }
          }
          *(uint4*)&lds_a[row * 40 + q * 8] = val;
        }
      }
    } else {
      // scalar path (first layer, CIN=3, K=48)
      for (int t = tid; t < MROWS * 32; t += 256) {
        const int row = t >> 5;
        const int kk  = t & 31;
        const int k   = k0 + kk;
        const int m   = mbase + row;
        const int ox  = m & wmask;
        const int oy  = (m >> p.wsh) & hmask;
        const int b   = m >> (p.wsh + p.hsh);
        float v = 0.0f;
        if (k < K) {
          const int ci = k % CIN;
          const int kc = k / CIN;
          const int kw = kc % KW;
          const int kh = kc / KW;
          const int iy = oy * STRIDE - PAD + kh;
          const int ix = ox * STRIDE - PAD + kw;
          if (iy >= 0 && iy < p.Hin && ix >= 0 && ix < p.Win) {
            float xv = (float)p.in[((size_t)(b * p.Hin + iy) * p.Win + ix) * CIN + ci];
            if (p.in_relu) xv = xv > 0.0f ? xv : 0.0f;
            v = xv;
          }
        }
        lds_a[row * 40 + kk] = (bf16)v;
      }
    }
    __syncthreads();

    // A fragment: two conflict-free b128 LDS reads per lane
    const int arow = wrow * 16 + r16;
    const v8bf a0 = *(const v8bf*)&lds_a[arow * 40 + (grp ? 8 : 0)];
    const v8bf a1 = *(const v8bf*)&lds_a[arow * 40 + 16 + (grp ? 8 : 0)];
    const v16bf afrag = __builtin_shufflevector(a0, a1, 0, 1, 2, 3, 4, 5, 6, 7,
                                                8, 9, 10, 11, 12, 13, 14, 15);
    // B fragment: contiguous 32B per lane, coalesced across the wave
    const v16bf bfrag = wf[(wfbase + c) * 32 + lane];
    if (c + 1 < NCH)
      __builtin_prefetch((const void*)&wf[(wfbase + c + 1) * 32 + lane], 0, 1);

    acc = __builtin_amdgcn_wmma_f32_16x16x32_bf16(false, afrag, false, bfrag,
                                                  (short)0, acc, false, false);
  }

  if (ncol < p.Cout) {
    const float bv = p.bias ? p.bias[ncol] : 0.0f;
#pragma unroll
    for (int r = 0; r < 8; ++r) {
      const int mm = mbase + wrow * 16 + r + grp * 8;  // D: M = r + 8*grp
      float val = acc[r] + bv;
      if (p.resid) val += (float)p.resid[(size_t)mm * p.Cout + ncol];
      if (p.out_relu) val = val > 0.0f ? val : 0.0f;
      if (p.out_bf) p.out_bf[(size_t)mm * p.Cout + ncol] = (bf16)val;
      if (p.out_f32) {
        if (p.out_mode == 2) {  // NCHW f32 (final decode)
          const int oxx = mm & wmask;
          const int oyy = (mm >> p.wsh) & hmask;
          const int bb  = mm >> (p.wsh + p.hsh);
          p.out_f32[(((size_t)bb * p.Cout + ncol) * p.Hout + oyy) * p.Wout + oxx] = val;
        } else {                // NHWC flat f32 (VQ input)
          p.out_f32[(size_t)mm * p.Cout + ncol] = val;
        }
      }
    }
  }
}

// OIHW f32 -> WMMA B-fragment order: [ntile][chunk][lane][16 elems] bf16
__global__ void pack_w_frag(const float* w, bf16* dst, int Cout, int Cin, int KH,
                            int KW, int K, int nchunks, int CoutPad) {
  const int idx = blockIdx.x * blockDim.x + threadIdx.x;
  const int total = (CoutPad / 16) * nchunks * 32 * 16;
  if (idx >= total) return;
  const int e    = idx & 15;
  const int lane = (idx >> 4) & 31;
  const int c    = (idx >> 9) % nchunks;
  const int nt   = (idx >> 9) / nchunks;
  const int v    = e >> 1;
  const int hh   = e & 1;
  const int k = c * 32 + 2 * (v & 3) + ((v >= 4) ? 16 : 0) + ((lane >= 16) ? 8 : 0) + hh;
  const int col = nt * 16 + (lane & 15);
  float val = 0.0f;
  if (col < Cout && k < K) {
    const int ci = k % Cin;
    const int kc = k / Cin;
    const int kw = kc % KW;
    const int kh = kc / KW;
    val = w[(((size_t)col * Cin + ci) * KH + kh) * KW + kw];
  }
  dst[idx] = (bf16)val;
}

__global__ void nchw_to_nhwc(const float* x, bf16* y, int B, int C, int H, int W) {
  const size_t idx = (size_t)blockIdx.x * blockDim.x + threadIdx.x;
  if (idx >= (size_t)B * C * H * W) return;
  const int c = (int)(idx % C);
  size_t t = idx / C;
  const int w = (int)(t % W); t /= W;
  const int h = (int)(t % H); t /= H;
  const int b = (int)t;
  y[idx] = (bf16)x[(((size_t)b * C + c) * H + h) * W + w];
}

__global__ void concat_c(const bf16* a, const bf16* b, bf16* o, int M, int C1, int C2) {
  const size_t idx = (size_t)blockIdx.x * blockDim.x + threadIdx.x;
  const int C = C1 + C2;
  if (idx >= (size_t)M * C) return;
  const int    c = (int)(idx % C);
  const size_t m = idx / C;
  o[idx] = (c < C1) ? a[m * C1 + c] : b[m * C2 + (c - C1)];
}

// VQ: per-row argmin over 512 codes (D=64), codebook tiled through LDS.
__global__ __launch_bounds__(128) void vq_kernel(const float* z, const float* embed,
                                                 bf16* q_out, float* ids_out,
                                                 float* diff_sum, float* counts, int M) {
  __shared__ float se[128 * 64];  // 32KB codebook tile
  __shared__ float sred[128];
  const int tid = threadIdx.x;
  const int row = blockIdx.x * 128 + tid;  // M is a multiple of 128
  float zv[64];
#pragma unroll
  for (int i = 0; i < 64; ++i) zv[i] = z[(size_t)row * 64 + i];
  float best = 3.4e38f;
  int bestid = 0;
  for (int c0 = 0; c0 < 512; c0 += 128) {
    for (int i = tid; i < 128 * 64; i += 128) se[i] = embed[(size_t)c0 * 64 + i];
    __syncthreads();
    for (int c = 0; c < 128; ++c) {
      float d = 0.0f;
#pragma unroll
      for (int i = 0; i < 64; ++i) {
        const float df = zv[i] - se[c * 64 + i];
        d += df * df;
      }
      if (d < best) { best = d; bestid = c0 + c; }
    }
    __syncthreads();
  }
  float s = 0.0f;
#pragma unroll
  for (int i = 0; i < 64; ++i) {
    const float qv = embed[(size_t)bestid * 64 + i];
    q_out[(size_t)row * 64 + i] = (bf16)qv;  // straight-through forward == q
    const float df = qv - zv[i];
    s += df * df;
  }
  ids_out[row] = (float)bestid;
  atomicAdd(&counts[bestid], 1.0f);
  sred[tid] = s;
  __syncthreads();
  for (int st = 64; st > 0; st >>= 1) {
    if (tid < st) sred[tid] += sred[tid + st];
    __syncthreads();
  }
  if (tid == 0) atomicAdd(diff_sum, sred[0]);
}

__global__ __launch_bounds__(512) void vq_finalize(const float* counts_t, const float* counts_b,
                                                   const float* dsum_t, const float* dsum_b,
                                                   float* out_diff, float* out_pt, float* out_pb,
                                                   int Mt, int Mb) {
  __shared__ float sh[512];
  const int t = threadIdx.x;
  const float at = counts_t[t] / (float)Mt;
  const float ab = counts_b[t] / (float)Mb;
  sh[t] = -at * __logf(at + 1e-10f);
  __syncthreads();
  for (int s = 256; s > 0; s >>= 1) { if (t < s) sh[t] += sh[t + s]; __syncthreads(); }
  const float Ht = sh[0];
  __syncthreads();
  sh[t] = -ab * __logf(ab + 1e-10f);
  __syncthreads();
  for (int s = 256; s > 0; s >>= 1) { if (t < s) sh[t] += sh[t + s]; __syncthreads(); }
  if (t == 0) {
    *out_pt   = __expf(Ht);
    *out_pb   = __expf(sh[0]);
    *out_diff = dsum_t[0] / ((float)Mt * 64.0f) + dsum_b[0] / ((float)Mb * 64.0f);
  }
}

// ---------------------------------------------------------------------------
template <int CIN, int KH, int KW, int STRIDE, int PAD, int TRN, int NW>
static void launch_conv(hipStream_t stream, const bf16* in, const bf16* wpk,
                        const float* bias, bf16* outb, float* outf, int outmode,
                        const bf16* resid, int Hin, int Win, int Hout, int Wout,
                        int Cout, int inRelu, int outRelu) {
  ConvP p;
  p.in = in; p.wpk = wpk; p.bias = bias;
  p.out_bf = outb; p.out_f32 = outf; p.resid = resid;
  p.Hin = Hin; p.Win = Win; p.Hout = Hout; p.Wout = Wout; p.Cout = Cout;
  p.wsh = __builtin_ctz((unsigned)Wout);
  p.hsh = __builtin_ctz((unsigned)Hout);
  p.in_relu = inRelu; p.out_relu = outRelu; p.out_mode = outmode;
  constexpr int MW = 8 / NW;
  const int CoutPad = (Cout + 15) & ~15;
  dim3 grid((unsigned)((16 * Hout * Wout) / (16 * MW)), (unsigned)(CoutPad / (16 * NW)));
  conv_wmma<CIN, KH, KW, STRIDE, PAD, TRN, NW><<<grid, 256, 0, stream>>>(p);
}

extern "C" void kernel_launch(void* const* d_in, const int* in_sizes, int n_in,
                              void* d_out, int out_size, void* d_ws, size_t ws_size,
                              hipStream_t stream) {
  (void)in_sizes; (void)n_in; (void)out_size; (void)ws_size;

  char* base = (char*)d_ws;
  size_t off = 0;
  auto alloc = [&](size_t bytes) -> void* {
    off = (off + 255) & ~(size_t)255;
    void* p = base + off;
    off += bytes;
    return p;
  };
  auto F = [&](int i) { return (const float*)d_in[i]; };

  auto packW = [&](int wi, int Cout, int Cin, int KH, int KW) -> bf16* {
    const int K = KH * KW * Cin;
    const int nchunks = (K + 31) / 32;
    const int CoutPad = (Cout + 15) & ~15;
    const int total = (CoutPad / 16) * nchunks * 32 * 16;
    bf16* dst = (bf16*)alloc((size_t)total * sizeof(bf16));
    pack_w_frag<<<(total + 255) / 256, 256, 0, stream>>>(F(wi), dst, Cout, Cin, KH, KW,
                                                         K, nchunks, CoutPad);
    return dst;
  };

  // ---- pack all weights (input indices per _param_specs order) ----
  bf16* W_eb0    = packW(1, 64, 3, 4, 4);
  bf16* W_eb1    = packW(3, 128, 64, 4, 4);
  bf16* W_eb2    = packW(5, 128, 128, 3, 3);
  bf16* W_ebr0w1 = packW(7, 32, 128, 3, 3);
  bf16* W_ebr0w2 = packW(9, 128, 32, 1, 1);
  bf16* W_ebr1w1 = packW(11, 32, 128, 3, 3);
  bf16* W_ebr1w2 = packW(13, 128, 32, 1, 1);
  bf16* W_et0    = packW(15, 64, 128, 4, 4);
  bf16* W_et1    = packW(17, 128, 64, 3, 3);
  bf16* W_etr0w1 = packW(19, 32, 128, 3, 3);
  bf16* W_etr0w2 = packW(21, 128, 32, 1, 1);
  bf16* W_etr1w1 = packW(23, 32, 128, 3, 3);
  bf16* W_etr1w2 = packW(25, 128, 32, 1, 1);
  bf16* W_qt     = packW(27, 64, 128, 1, 1);
  bf16* W_dt0    = packW(30, 128, 64, 3, 3);
  bf16* W_dtr0w1 = packW(32, 32, 128, 3, 3);
  bf16* W_dtr0w2 = packW(34, 128, 32, 1, 1);
  bf16* W_dtr1w1 = packW(36, 32, 128, 3, 3);
  bf16* W_dtr1w2 = packW(38, 128, 32, 1, 1);
  bf16* W_dtT    = packW(40, 64, 128, 4, 4);
  bf16* W_qb     = packW(42, 64, 192, 1, 1);
  bf16* W_up     = packW(45, 64, 64, 4, 4);
  bf16* W_d0     = packW(47, 128, 128, 3, 3);
  bf16* W_dr0w1  = packW(49, 32, 128, 3, 3);
  bf16* W_dr0w2  = packW(51, 128, 32, 1, 1);
  bf16* W_dr1w1  = packW(53, 32, 128, 3, 3);
  bf16* W_dr1w2  = packW(55, 128, 32, 1, 1);
  bf16* W_T1     = packW(57, 64, 128, 4, 4);
  bf16* W_T2     = packW(59, 3, 64, 4, 4);

  // ---- activation buffers ----
  bf16* xh   = (bf16*)alloc((size_t)16 * 256 * 256 * 3 * 2);
  bf16* h0   = (bf16*)alloc((size_t)16 * 128 * 128 * 64 * 2);
  bf16* bufA = (bf16*)alloc((size_t)16 * 64 * 64 * 128 * 2);  // h1; reused later as zb f32
  bf16* bufB = (bf16*)alloc((size_t)16 * 64 * 64 * 128 * 2);  // enc_b (in-place res stack)
  bf16* y64  = (bf16*)alloc((size_t)16 * 64 * 64 * 32 * 2);
  bf16* t0   = (bf16*)alloc((size_t)16 * 32 * 32 * 64 * 2);
  bf16* bufC = (bf16*)alloc((size_t)16 * 32 * 32 * 128 * 2);  // enc_t
  bf16* y32  = (bf16*)alloc((size_t)16 * 32 * 32 * 32 * 2);
  float* zt  = (float*)alloc((size_t)16 * 32 * 32 * 64 * 4);
  bf16* qt   = (bf16*)alloc((size_t)16 * 32 * 32 * 64 * 2);   // quant_t
  bf16* bufF = (bf16*)alloc((size_t)16 * 32 * 32 * 128 * 2);  // dec_t stage
  bf16* bufG = (bf16*)alloc((size_t)16 * 64 * 64 * 64 * 2);   // dec_t (upsampled)
  bf16* qb   = (bf16*)alloc((size_t)16 * 64 * 64 * 64 * 2);   // quant_b
  bf16* catb = (bf16*)alloc((size_t)16 * 64 * 64 * 192 * 2);
  bf16* up   = (bf16*)alloc((size_t)16 * 64 * 64 * 64 * 2);
  bf16* qcat = (bf16*)alloc((size_t)16 * 64 * 64 * 128 * 2);
  bf16* decs = (bf16*)alloc((size_t)16 * 64 * 64 * 128 * 2);
  bf16* dT1  = (bf16*)alloc((size_t)16 * 128 * 128 * 64 * 2);
  float* counts_t = (float*)alloc(512 * 4);
  float* counts_b = (float*)alloc(512 * 4);
  float* dsum     = (float*)alloc(2 * 4);
  float* zb = (float*)bufA;  // bufA dead after enc_b_w2 conv

  (void)hipMemsetAsync(counts_t, 0, 512 * 4, stream);
  (void)hipMemsetAsync(counts_b, 0, 512 * 4, stream);
  (void)hipMemsetAsync(dsum, 0, 2 * 4, stream);

  // ---- output layout (floats) ----
  float* out      = (float*)d_out;
  float* out_dec  = out;                    // [16,3,256,256] NCHW
  float* out_diff = out + 3145728;
  float* out_pt   = out + 3145729;
  float* out_pb   = out + 3145730;
  float* out_idt  = out + 3145731;          // [16,32,32]
  float* out_idb  = out + 3145731 + 16384;  // [16,64,64]

  auto res_stack = [&](bf16* h, bf16* ytmp, int HW,
                       const bf16* w1a, const float* b1a, const bf16* w2a, const float* b2a,
                       const bf16* w1b, const float* b1b, const bf16* w2b, const float* b2b,
                       int finalRelu) {
    launch_conv<128, 3, 3, 1, 1, 0, 2>(stream, h, w1a, b1a, ytmp, nullptr, 0, nullptr,
                                       HW, HW, HW, HW, 32, 1, 0);
    launch_conv<32, 1, 1, 1, 0, 0, 8>(stream, ytmp, w2a, b2a, h, nullptr, 0, h,
                                      HW, HW, HW, HW, 128, 1, 0);
    launch_conv<128, 3, 3, 1, 1, 0, 2>(stream, h, w1b, b1b, ytmp, nullptr, 0, nullptr,
                                       HW, HW, HW, HW, 32, 1, 0);
    launch_conv<32, 1, 1, 1, 0, 0, 8>(stream, ytmp, w2b, b2b, h, nullptr, 0, h,
                                      HW, HW, HW, HW, 128, 1, finalRelu);
  };

  // ---- bottom encoder ----
  nchw_to_nhwc<<<(16 * 3 * 256 * 256 + 255) / 256, 256, 0, stream>>>(F(0), xh, 16, 3, 256, 256);
  launch_conv<3, 4, 4, 2, 1, 0, 4>(stream, xh, W_eb0, F(2), h0, nullptr, 0, nullptr,
                                   256, 256, 128, 128, 64, 0, 1);
  launch_conv<64, 4, 4, 2, 1, 0, 8>(stream, h0, W_eb1, F(4), bufA, nullptr, 0, nullptr,
                                    128, 128, 64, 64, 128, 0, 1);
  launch_conv<128, 3, 3, 1, 1, 0, 8>(stream, bufA, W_eb2, F(6), bufB, nullptr, 0, nullptr,
                                     64, 64, 64, 64, 128, 0, 0);
  res_stack(bufB, y64, 64, W_ebr0w1, F(8), W_ebr0w2, F(10),
            W_ebr1w1, F(12), W_ebr1w2, F(14), 1);  // bufB = relu(enc_b)

  // ---- top encoder ----
  launch_conv<128, 4, 4, 2, 1, 0, 4>(stream, bufB, W_et0, F(16), t0, nullptr, 0, nullptr,
                                     64, 64, 32, 32, 64, 0, 1);
  launch_conv<64, 3, 3, 1, 1, 0, 8>(stream, t0, W_et1, F(18), bufC, nullptr, 0, nullptr,
                                    32, 32, 32, 32, 128, 0, 0);
  res_stack(bufC, y32, 32, W_etr0w1, F(20), W_etr0w2, F(22),
            W_etr1w1, F(24), W_etr1w2, F(26), 1);  // bufC = relu(enc_t)

  // ---- top quantization ----
  launch_conv<128, 1, 1, 1, 0, 0, 4>(stream, bufC, W_qt, F(28), nullptr, zt, 1, nullptr,
                                     32, 32, 32, 32, 64, 0, 0);
  vq_kernel<<<16384 / 128, 128, 0, stream>>>(zt, F(29), qt, out_idt, &dsum[0], counts_t, 16384);

  // ---- top decoder ----
  launch_conv<64, 3, 3, 1, 1, 0, 8>(stream, qt, W_dt0, F(31), bufF, nullptr, 0, nullptr,
                                    32, 32, 32, 32, 128, 0, 0);
  res_stack(bufF, y32, 32, W_dtr0w1, F(33), W_dtr0w2, F(35),
            W_dtr1w1, F(37), W_dtr1w2, F(39), 1);
  launch_conv<128, 4, 4, 1, 0, 1, 4>(stream, bufF, W_dtT, F(41), bufG, nullptr, 0, nullptr,
                                     32, 32, 64, 64, 64, 0, 0);

  // ---- bottom quantization ----
  concat_c<<<(16 * 64 * 64 * 192 + 255) / 256, 256, 0, stream>>>(bufG, bufB, catb,
                                                                 16 * 64 * 64, 64, 128);
  launch_conv<192, 1, 1, 1, 0, 0, 4>(stream, catb, W_qb, F(43), nullptr, zb, 1, nullptr,
                                     64, 64, 64, 64, 64, 0, 0);
  vq_kernel<<<65536 / 128, 128, 0, stream>>>(zb, F(44), qb, out_idb, &dsum[1], counts_b, 65536);

  // ---- decode ----
  launch_conv<64, 4, 4, 1, 0, 1, 4>(stream, qt, W_up, F(46), up, nullptr, 0, nullptr,
                                    32, 32, 64, 64, 64, 0, 0);
  concat_c<<<(16 * 64 * 64 * 128 + 255) / 256, 256, 0, stream>>>(up, qb, qcat,
                                                                 16 * 64 * 64, 64, 64);
  launch_conv<128, 3, 3, 1, 1, 0, 8>(stream, qcat, W_d0, F(48), decs, nullptr, 0, nullptr,
                                     64, 64, 64, 64, 128, 0, 0);
  res_stack(decs, y64, 64, W_dr0w1, F(50), W_dr0w2, F(52),
            W_dr1w1, F(54), W_dr1w2, F(56), 1);
  launch_conv<128, 4, 4, 1, 0, 1, 4>(stream, decs, W_T1, F(58), dT1, nullptr, 0, nullptr,
                                     64, 64, 128, 128, 64, 0, 1);
  launch_conv<64, 4, 4, 1, 0, 1, 1>(stream, dT1, W_T2, F(60), nullptr, out_dec, 2, nullptr,
                                    128, 128, 256, 256, 3, 0, 0);

  // ---- scalars: diff, perplexities ----
  vq_finalize<<<1, 512, 0, stream>>>(counts_t, counts_b, &dsum[0], &dsum[1],
                                     out_diff, out_pt, out_pb, 16384, 65536);
}
